// GAT_24919400251510
// MI455X (gfx1250) — matrix operand order repaired and checked by reading, not compiled, not run on hardware
//
#include <hip/hip_runtime.h>

#define NND   50000     // nodes
#define NE    800000    // raw edges
#define NET   850000    // edges + self loops
#define FDIM  128       // H*C = 4*32
#define NH    4
#define CH    32

typedef __bf16 v16bf __attribute__((ext_vector_type(16)));
typedef float  v8f   __attribute__((ext_vector_type(8)));
typedef int    v8i   __attribute__((ext_vector_type(8)));
typedef unsigned int u32x4 __attribute__((ext_vector_type(4)));
typedef int    i32x4 __attribute__((ext_vector_type(4)));
typedef int    i32x8 __attribute__((ext_vector_type(8)));

__device__ __forceinline__ unsigned short f2bf(float f) {
    unsigned u = __float_as_uint(f);
    u = (u + 0x7FFFu + ((u >> 16) & 1u)) >> 16;   // round-to-nearest-even
    return (unsigned short)u;
}
// order-preserving monotonic float<->uint map for atomic max
__device__ __forceinline__ unsigned fenc(float x) {
    unsigned u = __float_as_uint(x);
    return (u & 0x80000000u) ? ~u : (u | 0x80000000u);
}
__device__ __forceinline__ float fdec(unsigned u) {
    return __uint_as_float((u & 0x80000000u) ? (u & 0x7FFFFFFFu) : ~u);
}

// ---------------- conversions ----------------
__global__ void k_conv_x(const float* __restrict__ x, unsigned short* __restrict__ xb, int total) {
    int t = blockIdx.x * blockDim.x + threadIdx.x;
    if (t < total) xb[t] = f2bf(x[t]);
}

// W row-major [K=128][N=128] -> Wt bf16 [N][K] (so B-fragment K pairs are contiguous)
__global__ void k_conv_wt(const float* __restrict__ W, unsigned short* __restrict__ wt) {
    int t = blockIdx.x * blockDim.x + threadIdx.x;
    if (t >= FDIM * FDIM) return;
    int n = t >> 7, k = t & 127;
    wt[t] = f2bf(W[k * FDIM + n]);
}

// ---------------- WMMA GEMM: h[rows x 128] = xb[rows x 128] * W ----------------
// Wt staged into LDS once per block via Tensor Data Mover; one wave per 16-row
// stripe, 8 column tiles of 16, K-loop 4 x 32.
__global__ __launch_bounds__(256) void k_gemm(const unsigned short* __restrict__ xb,
                                              const unsigned short* __restrict__ wt,
                                              float* __restrict__ h, int rows) {
    __shared__ alignas(128) unsigned short ldsW[FDIM * FDIM];   // 32 KB of 320 KB WGP LDS

    if ((threadIdx.x >> 5) == 0) {
        // TDM descriptor: 1 "row" of 4096 x 8B elements = contiguous 32 KB, tile == tensor
        unsigned lds_base = (unsigned)(size_t)(void*)&ldsW[0];          // LDS byte offset (addr[31:0])
        unsigned long long ga = (unsigned long long)(size_t)(const void*)wt;
        u32x4 g0;
        g0[0] = 1u;                                           // count=1, is_restore=0, gather off
        g0[1] = lds_base;                                     // lds_addr [63:32]
        g0[2] = (unsigned)(ga & 0xFFFFFFFFull);               // global_addr [95:64]
        g0[3] = (unsigned)((ga >> 32) & 0x01FFFFFFull) | (2u << 30);  // global_addr[56:32] | type=2
        i32x8 g1;
        g1[0] = 0x30000;               // workgroup_mask=0 | data_size=3 (8B) | no pad/iterate
        g1[1] = (int)(4096u << 16);    // atomic_barrier_addr=0 | tensor_dim0[15:0]=4096
        g1[2] = (int)(1u << 16);       // tensor_dim0[31:16]=0 | tensor_dim1[15:0]=1
        g1[3] = (int)(4096u << 16);    // tensor_dim1[31:16]=0 | tile_dim0=4096
        g1[4] = 1;                     // tile_dim1=1 | tile_dim2=0
        g1[5] = 4096;                  // tensor_dim0_stride[31:0]
        g1[6] = 0;                     // tensor_dim0_stride[47:32]=0 | tensor_dim1_stride[15:0]=0
        g1[7] = 0;
        i32x4 z4 = {};
#if defined(__clang_major__) && (__clang_major__ >= 23)
        i32x8 z8 = {};
        __builtin_amdgcn_tensor_load_to_lds(g0, g1, z4, z4, z8, 0);
#else
        __builtin_amdgcn_tensor_load_to_lds(g0, g1, z4, z4, 0);
#endif
        __builtin_amdgcn_s_wait_tensorcnt(0);
    }
    __syncthreads();

    const int wave = threadIdx.x >> 5;
    const int lane = threadIdx.x & 31;
    const int m0 = (blockIdx.x * 8 + wave) * 16;
    if (m0 >= rows) return;              // after barrier; wave-uniform, EXEC all-ones for WMMA
    const int lhalf = lane >> 4;
    const int lmod  = lane & 15;

    v8f acc[8];
    v8f zero = {};
    #pragma unroll
    for (int t = 0; t < 8; ++t) acc[t] = zero;

    int m = m0 + lmod;
    if (m >= rows) m = rows - 1;
    const unsigned* arow = (const unsigned*)xb + (size_t)m * (FDIM / 2);
    const unsigned* blds = (const unsigned*)ldsW;

    #pragma unroll
    for (int kk = 0; kk < FDIM; kk += 32) {
        // A fragment 16x32 bf16: lanes 0-15 hold K {kk..kk+7, kk+16..kk+23}, lanes 16-31 offset +8
        union { v8i i; v16bf b; } A;
        #pragma unroll
        for (int v = 0; v < 8; ++v) {
            int kb = kk + ((v & 3) << 1) + ((v >> 2) << 4) + (lhalf << 3);
            A.i[v] = arow[kb >> 1];
        }
        #pragma unroll
        for (int t = 0; t < 8; ++t) {
            // B fragment 32x16 bf16 from LDS: lane -> column n = t*16+lmod,
            // lanes 0-15 K {kk..kk+7, kk+16..kk+23}, lanes 16-31 offset +8
            int base = (t * 16 + lmod) * (FDIM / 2) + (kk >> 1) + (lhalf << 2);
            union { v8i i; v16bf b; } B;
            #pragma unroll
            for (int v = 0; v < 8; ++v) B.i[v] = blds[base + (v & 3) + ((v >> 2) << 3)];
            acc[t] = __builtin_amdgcn_wmma_f32_16x16x32_bf16(
                false, A.b, false, B.b, (short)0, acc[t], false, false);
        }
    }
    // C/D layout: VGPR r -> row m0 + r + 8*lhalf, col = t*16 + lmod
    const int mbase = m0 + (lhalf << 3);
    #pragma unroll
    for (int t = 0; t < 8; ++t) {
        #pragma unroll
        for (int r = 0; r < 8; ++r) {
            int mm = mbase + r;
            if (mm < rows) h[(size_t)mm * FDIM + t * 16 + lmod] = acc[t][r];
        }
    }
}

// ---------------- per-node attention logits ----------------
__global__ void k_alpha(const float* __restrict__ h, const float* __restrict__ a_src,
                        const float* __restrict__ a_dst, float* __restrict__ as_,
                        float* __restrict__ ad_) {
    int t = blockIdx.x * blockDim.x + threadIdx.x;
    if (t >= NND * NH) return;
    int node = t >> 2, hd = t & 3;
    const float* hp = h + (size_t)node * FDIM + hd * CH;
    const float* sp = a_src + hd * CH;
    const float* dp = a_dst + hd * CH;
    float ss = 0.f, dd = 0.f;
    #pragma unroll
    for (int c = 0; c < CH; ++c) {
        float v = hp[c];
        ss = fmaf(v, sp[c], ss);
        dd = fmaf(v, dp[c], dd);
    }
    as_[t] = ss;
    ad_[t] = dd;
}

// ---------------- init: zero aggregation buffer + segment state ----------------
__global__ void k_init(float* __restrict__ out, unsigned* __restrict__ emax,
                       float* __restrict__ denom, int total) {
    int t = blockIdx.x * blockDim.x + threadIdx.x;
    if (t < total) out[t] = 0.f;
    if (t < NND * NH) { emax[t] = 0u; denom[t] = 0.f; }
}

__device__ __forceinline__ void edge_sd(int eid, const int* __restrict__ ei, int& s, int& d) {
    if (eid < NE) { s = ei[eid]; d = ei[NE + eid]; }
    else          { s = d = eid - NE; }          // self loops appended
}

// ---------------- segment max (monotonic-uint atomic max) ----------------
__global__ void k_edge_max(const int* __restrict__ ei, const float* __restrict__ as_,
                           const float* __restrict__ ad_, unsigned* __restrict__ emax) {
    int t = blockIdx.x * blockDim.x + threadIdx.x;
    if (t >= NET * NH) return;
    int eid = t >> 2, hd = t & 3;
    int s, d; edge_sd(eid, ei, s, d);
    float e = as_[s * NH + hd] + ad_[d * NH + hd];
    e = e > 0.f ? e : 0.2f * e;                  // leaky relu
    atomicMax(&emax[d * NH + hd], fenc(e));
}

// ---------------- exp + segment sum ----------------
__global__ void k_edge_exp(const int* __restrict__ ei, const float* __restrict__ as_,
                           const float* __restrict__ ad_, const unsigned* __restrict__ emax,
                           float* __restrict__ denom, float* __restrict__ eexp) {
    int t = blockIdx.x * blockDim.x + threadIdx.x;
    if (t >= NET * NH) return;
    int eid = t >> 2, hd = t & 3;
    int s, d; edge_sd(eid, ei, s, d);
    float e = as_[s * NH + hd] + ad_[d * NH + hd];
    e = e > 0.f ? e : 0.2f * e;
    float ex = __expf(e - fdec(emax[d * NH + hd]));
    eexp[t] = ex;
    atomicAdd(&denom[d * NH + hd], ex);
}

// ---------------- wave-per-edge weighted scatter-aggregate ----------------
__global__ __launch_bounds__(256) void k_aggregate(const int* __restrict__ ei,
                                                   const float* __restrict__ h,
                                                   const float* __restrict__ eexp,
                                                   const float* __restrict__ denom,
                                                   float* __restrict__ out) {
    int eid = blockIdx.x * 8 + (threadIdx.x >> 5);
    if (eid >= NET) return;
    int lane = threadIdx.x & 31;
    int s, d; edge_sd(eid, ei, s, d);
    int head = lane >> 3;                        // lane covers channels [lane*4, lane*4+4) -> one head
    float w = eexp[eid * NH + head] / (denom[d * NH + head] + 1e-16f);
    const float4 hv = *(const float4*)(h + (size_t)s * FDIM + lane * 4);
    float* op = out + (size_t)d * FDIM + lane * 4;
    atomicAdd(op + 0, hv.x * w);
    atomicAdd(op + 1, hv.y * w);
    atomicAdd(op + 2, hv.z * w);
    atomicAdd(op + 3, hv.w * w);
}

// ---------------- bias (+ELU) (+bf16 re-encode for next layer) ----------------
__global__ void k_epilogue(float* __restrict__ out, const float* __restrict__ bias,
                           unsigned short* __restrict__ bf_out, int do_elu, int total) {
    int t = blockIdx.x * blockDim.x + threadIdx.x;
    if (t >= total) return;
    float v = out[t] + bias[t & (FDIM - 1)];
    if (do_elu) v = v > 0.f ? v : expm1f(v);
    if (bf_out) bf_out[t] = f2bf(v);
    else        out[t] = v;
}

// ---------------- host-side layer driver ----------------
static void run_layer(const unsigned short* xb, const unsigned short* wt,
                      const float* a_src, const float* a_dst, const float* bias,
                      const int* ei, float* h, float* as_, float* ad_,
                      unsigned* emax, float* denom, float* eexp,
                      float* out, unsigned short* bf_out, int do_elu, hipStream_t s) {
    dim3 b(256);
    k_gemm     <<<dim3((NND + 127) / 128),            b, 0, s>>>(xb, wt, h, NND);
    k_alpha    <<<dim3((NND * NH + 255) / 256),       b, 0, s>>>(h, a_src, a_dst, as_, ad_);
    k_init     <<<dim3((NND * FDIM + 255) / 256),     b, 0, s>>>(out, emax, denom, NND * FDIM);
    k_edge_max <<<dim3((NET * NH + 255) / 256),       b, 0, s>>>(ei, as_, ad_, emax);
    k_edge_exp <<<dim3((NET * NH + 255) / 256),       b, 0, s>>>(ei, as_, ad_, emax, denom, eexp);
    k_aggregate<<<dim3((NET + 7) / 8),                b, 0, s>>>(ei, h, eexp, denom, out);
    k_epilogue <<<dim3((NND * FDIM + 255) / 256),     b, 0, s>>>(out, bias, bf_out, do_elu, NND * FDIM);
}

extern "C" void kernel_launch(void* const* d_in, const int* in_sizes, int n_in,
                              void* d_out, int out_size, void* d_ws, size_t ws_size,
                              hipStream_t stream) {
    (void)in_sizes; (void)n_in; (void)out_size; (void)ws_size;
    const float* x   = (const float*)d_in[0];
    const int*   ei  = (const int*)d_in[1];
    const float* W1  = (const float*)d_in[2];
    const float* a1s = (const float*)d_in[3];
    const float* a1d = (const float*)d_in[4];
    const float* b1  = (const float*)d_in[5];
    const float* W2  = (const float*)d_in[6];
    const float* a2s = (const float*)d_in[7];
    const float* a2d = (const float*)d_in[8];
    const float* b2  = (const float*)d_in[9];
    float* out = (float*)d_out;

    char* ws = (char*)d_ws;
    size_t off = 0;
    auto carve = [&](size_t bytes) -> char* {
        char* p = ws + off;
        off = (off + bytes + 255) & ~(size_t)255;
        return p;
    };
    unsigned short* xb   = (unsigned short*)carve((size_t)NND * FDIM * 2);
    unsigned short* wt1  = (unsigned short*)carve((size_t)FDIM * FDIM * 2);
    unsigned short* wt2  = (unsigned short*)carve((size_t)FDIM * FDIM * 2);
    float*          h    = (float*)carve((size_t)NND * FDIM * 4);
    float*          out1 = (float*)carve((size_t)NND * FDIM * 4);
    float*          as_  = (float*)carve((size_t)NND * NH * 4);
    float*          ad_  = (float*)carve((size_t)NND * NH * 4);
    unsigned*       emax = (unsigned*)carve((size_t)NND * NH * 4);
    float*          denom= (float*)carve((size_t)NND * NH * 4);
    float*          eexp = (float*)carve((size_t)NET * NH * 4);

    dim3 b(256);
    k_conv_x <<<dim3((NND * FDIM + 255) / 256), b, 0, stream>>>(x, xb, NND * FDIM);
    k_conv_wt<<<dim3((FDIM * FDIM + 255) / 256), b, 0, stream>>>(W1, wt1);
    k_conv_wt<<<dim3((FDIM * FDIM + 255) / 256), b, 0, stream>>>(W2, wt2);

    // layer 1: aggregate into out1, ELU, re-encode bf16 into xb (layer-2 GEMM input)
    run_layer(xb, wt1, a1s, a1d, b1, ei, h, as_, ad_, emax, denom, eexp, out1, xb, 1, stream);
    // layer 2: aggregate straight into d_out, bias only
    run_layer(xb, wt2, a2s, a2d, b2, ei, h, as_, ad_, emax, denom, eexp, out, nullptr, 0, stream);
}
